// ExpertCompoundTracker_1271310319887
// MI455X (gfx1250) — 8-bit matrix operand order repaired and checked by reading, compiled, not run on hardware
//
#include <hip/hip_runtime.h>
#include <hip/hip_bf16.h>

typedef __attribute__((ext_vector_type(8))) int v8i;

#define TOPK 4
#define NEXP 64
#define EMA_DECAY_F 0.99f


__global__ void ect_init_ws(int* ws) {
    int gid = blockIdx.x * blockDim.x + threadIdx.x;
    if (gid < NEXP + NEXP * NEXP) ws[gid] = 0;
}

__device__ __forceinline__ unsigned lds_off(const void* p) {
    return (unsigned)(uintptr_t)(__attribute__((address_space(3))) const void*)p;
}

__device__ __forceinline__ void async_g2l_b128(unsigned ldsByteOff, const void* gaddr) {
    asm volatile("global_load_async_to_lds_b128 %0, %1, off"
                 :: "v"(ldsByteOff), "v"(gaddr) : "memory");
}
__device__ __forceinline__ void wait_asynccnt0() {
    asm volatile("s_wait_asynccnt 0x0" ::: "memory");
}

__global__ __launch_bounds__(512)
void ect_gram_kernel(const int* __restrict__ idx, int* __restrict__ ws, int nTokens) {
    __shared__ __align__(16) int sRaw[2][64 * TOPK];
    __shared__ unsigned int sIdx4[64];
    __shared__ __align__(16) unsigned char sH[NEXP * 64];
    __shared__ int sCounts[NEXP];

    const int tid  = threadIdx.x;
    const int lane = tid & 31;
    const int wv   = tid >> 5;
    const int ti   = wv >> 2;
    const int tj   = wv & 3;
    const int h    = lane >> 4;
    const int l15  = lane & 15;

    if (tid < NEXP) sCounts[tid] = 0;
    __syncthreads();

    v8i acc = {};
    int countAcc = 0;

    const int eRowH = tid >> 3;
    const int tBase = (tid & 7) * 8;

    const int nChunks = (nTokens + 63) >> 6;
    const int G = gridDim.x;
    const unsigned rawBase = lds_off(&sRaw[0][0]);

    if (wv < 2) {
        const int ch0 = blockIdx.x;
        if (ch0 < nChunks) {
            const int t = (ch0 << 6) + tid;
            if (t < nTokens)
                async_g2l_b128(rawBase + (unsigned)tid * 16u, idx + (size_t)t * TOPK);
        }
    }

    int buf = 0;
    for (int ch = blockIdx.x; ch < nChunks; ch += G) {
        __syncthreads();
        if (wv < 2) {
            wait_asynccnt0();
            const int t = (ch << 6) + tid;
            unsigned w = 0xFFFFFFFFu;
            if (t < nTokens) {
                int4 q = *reinterpret_cast<const int4*>(&sRaw[buf][tid * TOPK]);
                w = (unsigned)(q.x & 63)         | ((unsigned)(q.y & 63) << 8) |
                    ((unsigned)(q.z & 63) << 16) | ((unsigned)(q.w & 63) << 24);
            }
            sIdx4[tid] = w;
            const int chN = ch + G;
            if (chN < nChunks) {
                const int tn = (chN << 6) + tid;
                if (tn < nTokens)
                    async_g2l_b128(rawBase + (unsigned)(buf ^ 1) * 1024u +
                                       (unsigned)tid * 16u,
                                   idx + (size_t)tn * TOPK);
            }
        }
        __syncthreads();
        {
            int cs = 0;
            #pragma unroll
            for (int dt = 0; dt < 8; ++dt) {
                const int t = tBase + dt;
                const unsigned w = sIdx4[t];
                int m = (int)((w & 0xffu)         == (unsigned)eRowH)
                      + (int)(((w >> 8)  & 0xffu) == (unsigned)eRowH)
                      + (int)(((w >> 16) & 0xffu) == (unsigned)eRowH)
                      + (int)((w >> 24)           == (unsigned)eRowH);
                sH[eRowH * 64 + t] = (unsigned char)m;
                cs += m;
            }
            countAcc += cs;
        }
        __syncthreads();
        const int* aRow = (const int*)(sH + (ti * 16 + l15) * 64);
        const int* bRow = (const int*)(sH + (tj * 16 + l15) * 64);
        v8i A, B;
        #pragma unroll
        for (int g = 0; g < 4; ++g) {
            const int off = (g * 16 + h * 8) >> 2;
            A[2 * g]     = aRow[off];
            A[2 * g + 1] = aRow[off + 1];
        }
        #pragma unroll
        for (int g = 0; g < 4; ++g) {
            const int off = (g * 16 + h * 8) >> 2;
            B[2 * g]     = bRow[off];
            B[2 * g + 1] = bRow[off + 1];
        }
        acc = __builtin_amdgcn_wmma_i32_16x16x64_iu8(false, A, false, B, acc,
                                                     false, false);
        buf ^= 1;
    }

    atomicAdd(&sCounts[eRowH], countAcc);
    __syncthreads();
    if (tid < NEXP) atomicAdd(&ws[tid], sCounts[tid]);

    int* gram = ws + NEXP;
    #pragma unroll
    for (int r = 0; r < 8; ++r) {
        const int row = ti * 16 + h * 8 + r;
        const int col = tj * 16 + l15;
        atomicAdd(&gram[row * NEXP + col], acc[r]);
    }
}

__global__ void ect_finalize(const int* __restrict__ ws,
                             const float* __restrict__ emaIn,
                             const float* __restrict__ coactIn,
                             float* __restrict__ out, int nTokens) {
    int gid = blockIdx.x * blockDim.x + threadIdx.x;
    if (gid < NEXP) {
        float load = (float)ws[gid] / (float)nTokens;
        out[gid] = emaIn[gid] * EMA_DECAY_F + load * (1.0f - EMA_DECAY_F);
    } else if (gid < NEXP + NEXP * NEXP) {
        int idx = gid - NEXP;
        int r = idx >> 6;
        int c = idx & 63;
        int v = ws[NEXP + idx];
        if (r == c) v -= ws[r];
        out[gid] = coactIn[idx] + (float)v;
    }
}

extern "C" void kernel_launch(void* const* d_in, const int* in_sizes, int n_in,
                              void* d_out, int out_size, void* d_ws, size_t ws_size,
                              hipStream_t stream) {
    const int*   expertIdx = (const int*)d_in[0];
    const float* emaIn     = (const float*)d_in[2];
    const float* coactIn   = (const float*)d_in[3];
    float* out = (float*)d_out;

    const int nTokens = in_sizes[0] / TOPK;
    const int wsElems = NEXP + NEXP * NEXP;

    ect_init_ws<<<(wsElems + 255) / 256, 256, 0, stream>>>((int*)d_ws);
    ect_gram_kernel<<<256, 512, 0, stream>>>(expertIdx, (int*)d_ws, nTokens);
    ect_finalize<<<(wsElems + 255) / 256, 256, 0, stream>>>((const int*)d_ws,
                                                            emaIn, coactIn, out,
                                                            nTokens);
}
